// MissingAwareSigMamba_25967372271867
// MI455X (gfx1250) — matrix operand order repaired and checked by reading, not compiled, not run on hardware
//
#include <hip/hip_runtime.h>
#include <hip/hip_fp16.h>

typedef _Float16 half_t;
typedef _Float16 v16h __attribute__((ext_vector_type(16)));
typedef float    v8f  __attribute__((ext_vector_type(8)));
typedef int      v4i  __attribute__((__vector_size__(4 * sizeof(int))));

#define B_   16
#define T_   336
#define C_   21
#define TP   324
#define NB   (B_*C_)     // 336
#define M2_  (B_*TP)     // 5184
#define DI_  256
#define DS_  16
#define DM_  128
#define KF   352         // fusion K padded (336 -> 352)
#define XN   64          // x_proj N padded (40 -> 64, so N tiles 64-wide)
#define SIGCH 584
#define PLNV 2016        // 96*21

// ---------------------------------------------------------------------------
// Async global -> LDS copy (CDNA5 GLOBAL_LOAD_ASYNC_TO_LDS_B128, ASYNCcnt),
// with a synchronous LDS-staging fallback if the builtin is not declared.
// Builtin signature (from compiler diagnostic): arg0 = addrspace(1) v4i* src,
// arg1 = addrspace(3) v4i* dst, then two immediate ints (offset, cpol).
// ---------------------------------------------------------------------------
#if defined(__has_builtin)
# if __has_builtin(__builtin_amdgcn_global_load_async_to_lds_b128)
#  define HAVE_ASYNC_LDS 1
# endif
#endif
#ifndef HAVE_ASYNC_LDS
# define HAVE_ASYNC_LDS 0
#endif

__device__ __forceinline__ void cp_b128_to_lds(half_t* ldsDst, const half_t* gSrc) {
#if HAVE_ASYNC_LDS
    __builtin_amdgcn_global_load_async_to_lds_b128(
        (__attribute__((address_space(1))) v4i*)(uintptr_t)gSrc,
        (__attribute__((address_space(3))) v4i*)ldsDst,
        0, 0);
#else
    *(float4*)ldsDst = *(const float4*)gSrc;
#endif
}

__device__ __forceinline__ void async_wait0() {
#if HAVE_ASYNC_LDS
# if __has_builtin(__builtin_amdgcn_s_wait_asynccnt)
    __builtin_amdgcn_s_wait_asynccnt(0);
# else
    asm volatile("s_wait_asynccnt 0" ::: "memory");
# endif
#endif
}

// ---------------------------------------------------------------------------
// Prep: f32 -> f16 weight conversion with zero padding
// ---------------------------------------------------------------------------
__global__ void cvt_pad_kernel(const float* __restrict__ src, half_t* __restrict__ dst,
                               int rows, int cols, int rowsSrc, int colsSrc) {
    int gid = blockIdx.x * blockDim.x + threadIdx.x;
    if (gid >= rows * cols) return;
    int r = gid / cols, c = gid % cols;
    float v = (r < rowsSrc && c < colsSrc) ? src[r * colsSrc + c] : 0.0f;
    dst[gid] = (half_t)v;
}

__global__ void zero_half_kernel(half_t* __restrict__ p, int n) {
    int gid = blockIdx.x * blockDim.x + threadIdx.x;
    if (gid < n) p[gid] = (half_t)0.0f;
}

// ---------------------------------------------------------------------------
// Stage 1: TCN (4 dilated convs, dil=2) + varproj(32->8), fused per (n,t)
// feat layout: [(n*TP + t)*8 + d]
// ---------------------------------------------------------------------------
__global__ void tcn_kernel(const float* __restrict__ x, const float* __restrict__ mk,
                           const float* __restrict__ dlt,
                           const float* __restrict__ w2, const float* __restrict__ b2,
                           const float* __restrict__ w3, const float* __restrict__ b3,
                           const float* __restrict__ w6, const float* __restrict__ b6,
                           const float* __restrict__ w7, const float* __restrict__ b7,
                           const float* __restrict__ vw, const float* __restrict__ vb,
                           float* __restrict__ feat) {
    int gid = blockIdx.x * blockDim.x + threadIdx.x;
    if (gid >= NB * TP) return;
    int n = gid / TP, t = gid % TP;
    int b = n / C_, c = n % C_;

    float conv[32];
    const float* ws[4] = { w2, w3, w6, w7 };
    const float* bs[4] = { b2, b3, b6, b7 };
    const int    Ks[4] = { 2, 3, 6, 7 };
    #pragma unroll
    for (int s = 0; s < 4; ++s) {
        const int K = Ks[s];
        const int base = t + 14 - 2 * K;   // output index (Lk-minT+t) -> input start
        #pragma unroll
        for (int o = 0; o < 8; ++o) {
            float acc = bs[s][o];
            #pragma unroll
            for (int ci = 0; ci < 3; ++ci) {
                const float* src = (ci == 0) ? x : ((ci == 1) ? mk : dlt);
                #pragma unroll
                for (int k = 0; k < 7; ++k) {
                    if (k < K)
                        acc += ws[s][(o * 3 + ci) * K + k] *
                               src[((size_t)b * T_ + base + 2 * k) * C_ + c];
                }
            }
            conv[s * 8 + o] = acc;
        }
    }
    float* fo = feat + (size_t)gid * 8;
    #pragma unroll
    for (int d = 0; d < 8; ++d) {
        float a = vb[d];
        #pragma unroll
        for (int j = 0; j < 32; ++j) a += vw[d * 32 + j] * conv[j];
        fo[d] = a;
    }
}

// ---------------------------------------------------------------------------
// Stage 2: depth-3 signature (one wave32 per window) fused with sigproj 584->8
// Lane l owns S2/S3 flat rows l and l+32; S1[l&7] distributed on lanes; v via shfl.
// Writes sigout (f32, [b][t][c][8]) and fin[:,0:168] (f16 fusion input).
// ---------------------------------------------------------------------------
__global__ __launch_bounds__(128) void sig_kernel(const float* __restrict__ feat,
                                                  const float* __restrict__ sw,
                                                  const float* __restrict__ sb,
                                                  float* __restrict__ sigout,
                                                  half_t* __restrict__ fin) {
    __shared__ float Wl[8 * SIGCH];
    for (int i = threadIdx.x; i < 8 * SIGCH; i += 128) Wl[i] = sw[i];
    __syncthreads();

    const int wid = blockIdx.x * 4 + (threadIdx.x >> 5);   // window id (exact fit)
    const int l   = threadIdx.x & 31;
    const int n = wid / TP, t = wid % TP;
    const float* fbase = feat + (size_t)n * TP * 8;

    // Load the 8 window points (64 floats) distributed: lane l holds flats l, l+32.
    const int p0 = l >> 3, d0 = l & 7;
    int ta = t - 7 + p0;     if (ta < 0) ta = 0;
    int tb = t - 7 + p0 + 4; if (tb < 0) tb = 0;
    const float w0 = fbase[ta * 8 + d0];
    const float w1 = fbase[tb * 8 + d0];

    const int i0 = l >> 3;                 // S2 pair a = (i0, j0), pair b = (i0+4, j0)
    float s1l = 0.0f;                      // this lane's S1[l & 7]
    float s2a = 0.0f, s2b = 0.0f;
    float s3a[8], s3b[8];
    #pragma unroll
    for (int k = 0; k < 8; ++k) { s3a[k] = 0.0f; s3b[k] = 0.0f; }

    #pragma unroll
    for (int s = 0; s < 7; ++s) {
        float v[8];
        float vdist = 0.0f;                // v[l & 7], kept without dynamic indexing
        #pragma unroll
        for (int d = 0; d < 8; ++d) {
            float hi = (s + 1 < 4) ? __shfl(w0, (s + 1) * 8 + d, 32)
                                   : __shfl(w1, (s - 3) * 8 + d, 32);
            float lo = (s < 4)     ? __shfl(w0, s * 8 + d, 32)
                                   : __shfl(w1, (s - 4) * 8 + d, 32);
            v[d] = hi - lo;
            if (d == (l & 7)) vdist = v[d];
        }
        const float s1_i0 = __shfl(s1l, i0, 32);
        const float s1_i1 = __shfl(s1l, i0 + 4, 32);
        const float vj  = vdist;                 // v[j0] since j0 == (l & 7)
        const float vi0 = __shfl(vdist, i0, 32);
        const float vi1 = __shfl(vdist, i0 + 4, 32);
        // S3[i,j,k] += v_k * ( S2_ij + v_j*(0.5*S1_i + v_i/6) )
        const float coefA = s2a + vj * (0.5f * s1_i0 + vi0 * (1.0f / 6.0f));
        const float coefB = s2b + vj * (0.5f * s1_i1 + vi1 * (1.0f / 6.0f));
        #pragma unroll
        for (int k = 0; k < 8; ++k) { s3a[k] += coefA * v[k]; s3b[k] += coefB * v[k]; }
        // S2 += v_j*(S1_i + 0.5*v_i);  S1 += v
        s2a += vj * (s1_i0 + 0.5f * vi0);
        s2b += vj * (s1_i1 + 0.5f * vi1);
        s1l += vdist;
    }

    // sigproj: out[o] = b[o] + sum_c W[o][c]*sig[c];  sig = [S1(8) | S2(64) | S3(512)]
    float acc[8];
    #pragma unroll
    for (int o = 0; o < 8; ++o) {
        const float* W = Wl + o * SIGCH;
        float a = 0.0f;
        if (l < 8) a += W[l] * s1l;
        a += W[8 + l] * s2a + W[8 + 32 + l] * s2b;
        #pragma unroll
        for (int k = 0; k < 8; ++k)
            a += W[72 + l * 8 + k] * s3a[k] + W[72 + (l + 32) * 8 + k] * s3b[k];
        #pragma unroll
        for (int off = 16; off > 0; off >>= 1) a += __shfl_xor(a, off, 32);
        acc[o] = a;
    }
    if (l == 0) {
        const int b = n / C_, c = n % C_;
        const size_t m2 = (size_t)b * TP + t;
        float*  so = sigout + (m2 * C_ + c) * 8;
        half_t* fo = fin + m2 * KF + c * 8;
        #pragma unroll
        for (int o = 0; o < 8; ++o) {
            float r = acc[o] + sb[o];
            so[o] = r;
            fo[o] = (half_t)r;
        }
    }
}

// ---------------------------------------------------------------------------
// Stage 3: MHA over the 21 variables (E=8, 4 heads, hd=2). One 32-thread block
// per (b,t); writes f16 output into fin[:, 168:336].
// ---------------------------------------------------------------------------
__global__ __launch_bounds__(32) void mha_kernel(const float* __restrict__ sigout,
                                                 const float* __restrict__ inw,
                                                 const float* __restrict__ inb,
                                                 const float* __restrict__ outw,
                                                 const float* __restrict__ outb,
                                                 half_t* __restrict__ fin) {
    const int m2 = blockIdx.x;         // 0..5183
    const int c  = threadIdx.x;
    __shared__ float sq[21][24];       // per-token q(8)|k(8)|v(8)

    if (c < 21) {
        float tok[8];
        #pragma unroll
        for (int d = 0; d < 8; ++d) tok[d] = sigout[((size_t)m2 * C_ + c) * 8 + d];
        #pragma unroll
        for (int j = 0; j < 24; ++j) {
            float s = inb[j];
            #pragma unroll
            for (int d = 0; d < 8; ++d) s += inw[j * 8 + d] * tok[d];
            sq[c][j] = s;
        }
    }
    __syncthreads();
    if (c < 21) {
        float o8[8];
        #pragma unroll
        for (int h = 0; h < 4; ++h) {
            const float q0 = sq[c][2 * h], q1 = sq[c][2 * h + 1];
            float sc[21];
            float mx = -1e30f;
            #pragma unroll
            for (int j = 0; j < 21; ++j) {
                float s = (q0 * sq[j][8 + 2 * h] + q1 * sq[j][9 + 2 * h]) * 0.70710678f;
                sc[j] = s;
                mx = fmaxf(mx, s);
            }
            float sum = 0.0f;
            #pragma unroll
            for (int j = 0; j < 21; ++j) { sc[j] = __expf(sc[j] - mx); sum += sc[j]; }
            const float inv = 1.0f / sum;
            float a0 = 0.0f, a1 = 0.0f;
            #pragma unroll
            for (int j = 0; j < 21; ++j) {
                float wgt = sc[j] * inv;
                a0 += wgt * sq[j][16 + 2 * h];
                a1 += wgt * sq[j][17 + 2 * h];
            }
            o8[2 * h] = a0; o8[2 * h + 1] = a1;
        }
        half_t* fo = fin + (size_t)m2 * KF + 168 + c * 8;
        #pragma unroll
        for (int d = 0; d < 8; ++d) {
            float a = outb[d];
            #pragma unroll
            for (int e = 0; e < 8; ++e) a += outw[d * 8 + e] * o8[e];
            fo[d] = (half_t)a;
        }
    }
}

// ---------------------------------------------------------------------------
// WMMA GEMM v2: C[M x Npad] = A[M x K] * Bw^T (+bias); Bw is (Npad x K) f16.
// Block = 128 threads (4 waves) computing a 64x64 output tile:
//   wave w -> rows [m0+16w, m0+16w+16), all four 16-wide N subtiles.
// Per K-step the 64x32 B tile is staged once into LDS (async global->LDS when
// available) and shared by all 4 waves; each wave reuses one A fragment across
// 4 WMMAs. A and B fragments share contiguous k = 16*half + s.
// ---------------------------------------------------------------------------
__global__ __launch_bounds__(128) void wmma_gemm_kernel(const half_t* __restrict__ A, int lda,
                                                        const half_t* __restrict__ Bw, int ldb,
                                                        float* __restrict__ Cf,
                                                        half_t* __restrict__ Ch, int ldc,
                                                        const float* __restrict__ bias, int K) {
    __shared__ __align__(32) half_t ldsB[64 * 32];    // 64 weight rows x 32 k (4 KB)
    const int tid  = threadIdx.x;
    const int wave = tid >> 5;
    const int l    = tid & 31;
    const int m0 = blockIdx.x * 64 + wave * 16;
    const int n0 = blockIdx.y * 64;
    const int half_id = l >> 4;
    const int lcol = l & 15;
    const int arow = m0 + lcol;          // A row for this lane

    v8f c[4];
    #pragma unroll
    for (int nt = 0; nt < 4; ++nt) {
        const float binit = bias ? bias[n0 + nt * 16 + lcol] : 0.0f;
        #pragma unroll
        for (int r = 0; r < 8; ++r) c[nt][r] = binit;
    }

    for (int k0 = 0; k0 < K; k0 += 32) {
        // Stage B tile: 256 x 16B transfers, 2 per thread.
        #pragma unroll
        for (int i = 0; i < 2; ++i) {
            const int idx = tid + 128 * i;           // 0..255
            const int row = idx >> 2;                // 0..63
            const int seg = idx & 3;                 // 16-byte segment in the row
            cp_b128_to_lds(&ldsB[row * 32 + seg * 8],
                           Bw + (size_t)(n0 + row) * ldb + k0 + seg * 8);
        }
        async_wait0();
        __syncthreads();

        // A fragment (16x32) direct from global, contiguous k = 16*half + s (matches B).
        v16h a;
        const half_t* ap = A + (size_t)arow * lda + k0;
        #pragma unroll
        for (int v = 0; v < 8; ++v) {
            const int kk = 2 * v + 16 * half_id;
            a[2 * v]     = ap[kk];
            a[2 * v + 1] = ap[kk + 1];
        }
        // 4 N subtiles reuse the A fragment; B fragments from LDS (contiguous 32B).
        #pragma unroll
        for (int nt = 0; nt < 4; ++nt) {
            const v16h b = *reinterpret_cast<const v16h*>(
                &ldsB[(nt * 16 + lcol) * 32 + 16 * half_id]);
            c[nt] = __builtin_amdgcn_wmma_f32_16x16x32_f16(false, a, false, b, (short)0,
                                                           c[nt], false, false);
        }
        __syncthreads();
    }

    #pragma unroll
    for (int nt = 0; nt < 4; ++nt) {
        #pragma unroll
        for (int r = 0; r < 8; ++r) {
            const int row = m0 + r + 8 * half_id;
            const size_t off = (size_t)row * ldc + n0 + nt * 16 + lcol;
            Cf[off] = c[nt][r];
            if (Ch) Ch[off] = (half_t)c[nt][r];
        }
    }
}

// ---------------------------------------------------------------------------
// Stage 5: mamba causal depthwise conv (K=4, left pad 3) + silu
// ---------------------------------------------------------------------------
__global__ void conv_silu_kernel(const float* __restrict__ xz, const float* __restrict__ cw,
                                 const float* __restrict__ cb,
                                 float* __restrict__ xma, half_t* __restrict__ xmh) {
    int gid = blockIdx.x * blockDim.x + threadIdx.x;
    if (gid >= M2_ * DI_) return;
    const int m2 = gid / DI_, d = gid % DI_;
    const int b = m2 / TP, t = m2 % TP;
    float acc = cb[d];
    #pragma unroll
    for (int k = 0; k < 4; ++k) {
        const int tt = t - 3 + k;
        if (tt >= 0) acc += cw[d * 4 + k] * xz[((size_t)b * TP + tt) * 512 + d];
    }
    const float s = acc / (1.0f + __expf(-acc));
    xma[gid] = s;
    xmh[gid] = (half_t)s;
}

// ---------------------------------------------------------------------------
// Stage 6: dt projection (8 -> 256) + softplus
// ---------------------------------------------------------------------------
__global__ void dt_kernel(const float* __restrict__ dbl, const float* __restrict__ dtw,
                          const float* __restrict__ dtb, float* __restrict__ dl) {
    int gid = blockIdx.x * blockDim.x + threadIdx.x;
    if (gid >= M2_ * DI_) return;
    const int m2 = gid / DI_, d = gid % DI_;
    float a = dtb[d];
    #pragma unroll
    for (int r = 0; r < 8; ++r) a += dbl[(size_t)m2 * XN + r] * dtw[d * 8 + r];
    dl[gid] = (a > 20.0f) ? a : log1pf(__expf(a));
}

// ---------------------------------------------------------------------------
// Stage 7: SSM scan over time; only the last-step y is needed downstream.
// grid = B, block = DI; h[16] in registers, -exp(A_log) cached in LDS.
// ---------------------------------------------------------------------------
__global__ __launch_bounds__(256) void scan_kernel(const float* __restrict__ dbl,
                                                   const float* __restrict__ dl,
                                                   const float* __restrict__ xma,
                                                   const float* __restrict__ Alog,
                                                   float* __restrict__ ylast) {
    __shared__ float As[DI_ * DS_];
    __shared__ float bc[DS_], cc[DS_];
    const int b = blockIdx.x, d = threadIdx.x;
    for (int i = d; i < DI_ * DS_; i += 256) As[i] = -__expf(Alog[i]);
    __syncthreads();
    float h[DS_];
    #pragma unroll
    for (int s = 0; s < DS_; ++s) h[s] = 0.0f;
    float y = 0.0f;
    for (int t = 0; t < TP; ++t) {
        const size_t m2 = (size_t)b * TP + t;
        if (d < 16)      bc[d] = dbl[m2 * XN + 8 + d];
        else if (d < 32) cc[d - 16] = dbl[m2 * XN + 24 + (d - 16)];
        __syncthreads();
        const float dv = dl[m2 * DI_ + d];
        const float xv = xma[m2 * DI_ + d];
        const float dx = dv * xv;
        #pragma unroll
        for (int s = 0; s < DS_; ++s)
            h[s] = h[s] * __expf(dv * As[d * DS_ + s]) + dx * bc[s];
        if (t == TP - 1) {
            y = 0.0f;
            #pragma unroll
            for (int s = 0; s < DS_; ++s) y += h[s] * cc[s];
        }
        __syncthreads();
    }
    ylast[b * DI_ + d] = y;
}

// ---------------------------------------------------------------------------
// Stage 8: gate + out_proj + head (last timestep only)
// ---------------------------------------------------------------------------
__global__ void ycomb_kernel(const float* __restrict__ ylast, const float* __restrict__ xz,
                             const float* __restrict__ xma, const float* __restrict__ mD,
                             float* __restrict__ yc) {
    int gid = blockIdx.x * blockDim.x + threadIdx.x;
    if (gid >= B_ * DI_) return;
    const int b = gid / DI_, d = gid % DI_;
    const size_t m2 = (size_t)b * TP + (TP - 1);
    const float z  = xz[m2 * 512 + 256 + d];
    const float sz = z / (1.0f + __expf(-z));
    yc[gid] = (ylast[gid] + mD[d] * xma[m2 * DI_ + d]) * sz;
}

__global__ void mo_kernel(const float* __restrict__ yc, const float* __restrict__ mow,
                          float* __restrict__ mo) {
    int gid = blockIdx.x * blockDim.x + threadIdx.x;
    if (gid >= B_ * DM_) return;
    const int b = gid / DM_, j = gid % DM_;
    float a = 0.0f;
    for (int d = 0; d < DI_; ++d) a += yc[b * DI_ + d] * mow[j * DI_ + d];
    mo[gid] = a;
}

__global__ void pred_kernel(const float* __restrict__ mo, const float* __restrict__ hw,
                            const float* __restrict__ hb, float* __restrict__ out) {
    int gid = blockIdx.x * blockDim.x + threadIdx.x;
    if (gid >= B_ * PLNV) return;
    const int b = gid / PLNV, p = gid % PLNV;
    float a = hb[p];
    for (int j = 0; j < DM_; ++j) a += mo[b * DM_ + j] * hw[p * DM_ + j];
    out[gid] = a;
}

// ---------------------------------------------------------------------------
extern "C" void kernel_launch(void* const* d_in, const int* in_sizes, int n_in,
                              void* d_out, int out_size, void* d_ws, size_t ws_size,
                              hipStream_t stream) {
    const float* x    = (const float*)d_in[0];
    const float* mk   = (const float*)d_in[1];
    const float* dlt  = (const float*)d_in[2];
    const float* w2   = (const float*)d_in[3];
    const float* b2   = (const float*)d_in[4];
    const float* w3   = (const float*)d_in[5];
    const float* b3   = (const float*)d_in[6];
    const float* w6   = (const float*)d_in[7];
    const float* b6   = (const float*)d_in[8];
    const float* w7   = (const float*)d_in[9];
    const float* b7   = (const float*)d_in[10];
    const float* vw   = (const float*)d_in[11];
    const float* vb   = (const float*)d_in[12];
    const float* sw   = (const float*)d_in[13];
    const float* sb   = (const float*)d_in[14];
    const float* aiw  = (const float*)d_in[15];
    const float* aib  = (const float*)d_in[16];
    const float* aow  = (const float*)d_in[17];
    const float* aob  = (const float*)d_in[18];
    const float* fw   = (const float*)d_in[19];
    const float* fb   = (const float*)d_in[20];
    const float* minw = (const float*)d_in[21];
    const float* mcw  = (const float*)d_in[22];
    const float* mcb  = (const float*)d_in[23];
    const float* mxw  = (const float*)d_in[24];
    const float* mdtw = (const float*)d_in[25];
    const float* mdtb = (const float*)d_in[26];
    const float* mAlog= (const float*)d_in[27];
    const float* mD   = (const float*)d_in[28];
    const float* mow  = (const float*)d_in[29];
    const float* hw   = (const float*)d_in[30];
    const float* hb   = (const float*)d_in[31];
    float* out = (float*)d_out;

    // Bump allocator over the workspace (total ~40 MB).
    char* wsp = (char*)d_ws;
    auto alloc = [&](size_t bytes) -> void* {
        void* p = (void*)wsp;
        wsp += (bytes + 255) & ~(size_t)255;
        return p;
    };
    float*  feat    = (float*)alloc((size_t)NB * TP * 8 * 4);
    float*  sigout  = (float*)alloc((size_t)M2_ * C_ * 8 * 4);
    half_t* fin     = (half_t*)alloc((size_t)M2_ * KF * 2);
    half_t* whF     = (half_t*)alloc((size_t)DM_ * KF * 2);
    half_t* whIn    = (half_t*)alloc((size_t)512 * DM_ * 2);
    half_t* whX     = (half_t*)alloc((size_t)XN * DI_ * 2);
    float*  fuse    = (float*)alloc((size_t)M2_ * DM_ * 4);
    half_t* fuseH   = (half_t*)alloc((size_t)M2_ * DM_ * 2);
    float*  xz      = (float*)alloc((size_t)M2_ * 512 * 4);
    float*  xma     = (float*)alloc((size_t)M2_ * DI_ * 4);
    half_t* xmh     = (half_t*)alloc((size_t)M2_ * DI_ * 2);
    float*  dbl     = (float*)alloc((size_t)M2_ * XN * 4);
    float*  dl      = (float*)alloc((size_t)M2_ * DI_ * 4);
    float*  ylast   = (float*)alloc((size_t)B_ * DI_ * 4);
    float*  yc      = (float*)alloc((size_t)B_ * DI_ * 4);
    float*  mo      = (float*)alloc((size_t)B_ * DM_ * 4);

    const int TB = 256;
    auto cdiv = [](int a, int b) { return (a + b - 1) / b; };

    // --- prep: weight conversion + fin padding ---
    cvt_pad_kernel<<<cdiv(DM_ * KF, TB), TB, 0, stream>>>(fw, whF, DM_, KF, DM_, 336);
    cvt_pad_kernel<<<cdiv(512 * DM_, TB), TB, 0, stream>>>(minw, whIn, 512, DM_, 512, DM_);
    cvt_pad_kernel<<<cdiv(XN * DI_, TB), TB, 0, stream>>>(mxw, whX, XN, DI_, 40, DI_);
    zero_half_kernel<<<cdiv(M2_ * KF, TB), TB, 0, stream>>>(fin, M2_ * KF);

    // --- stage 1: TCN + varproj ---
    tcn_kernel<<<cdiv(NB * TP, TB), TB, 0, stream>>>(x, mk, dlt, w2, b2, w3, b3, w6, b6,
                                                     w7, b7, vw, vb, feat);

    // --- stage 2: signature + sigproj (wave per window) ---
    sig_kernel<<<(NB * TP) / 4, 128, 0, stream>>>(feat, sw, sb, sigout, fin);

    // --- stage 3: MHA over variables ---
    mha_kernel<<<M2_, 32, 0, stream>>>(sigout, aiw, aib, aow, aob, fin);

    // --- stage 4: fusion GEMM (5184 x 352 -> 128), WMMA, bias, also f16 copy ---
    wmma_gemm_kernel<<<dim3(M2_ / 64, DM_ / 64), 128, 0, stream>>>(
        fin, KF, whF, KF, fuse, fuseH, DM_, fb, KF);

    // --- mamba in_proj GEMM (5184 x 128 -> 512), WMMA ---
    wmma_gemm_kernel<<<dim3(M2_ / 64, 512 / 64), 128, 0, stream>>>(
        fuseH, DM_, whIn, DM_, xz, (half_t*)nullptr, 512, (const float*)nullptr, DM_);

    // --- stage 5: depthwise causal conv + silu ---
    conv_silu_kernel<<<cdiv(M2_ * DI_, TB), TB, 0, stream>>>(xz, mcw, mcb, xma, xmh);

    // --- x_proj GEMM (5184 x 256 -> 64 padded), WMMA ---
    wmma_gemm_kernel<<<dim3(M2_ / 64, XN / 64), 128, 0, stream>>>(
        xmh, DI_, whX, DI_, dbl, (half_t*)nullptr, XN, (const float*)nullptr, DI_);

    // --- stage 6: dt proj + softplus ---
    dt_kernel<<<cdiv(M2_ * DI_, TB), TB, 0, stream>>>(dbl, mdtw, mdtb, dl);

    // --- stage 7: SSM scan ---
    scan_kernel<<<B_, DI_, 0, stream>>>(dbl, dl, xma, mAlog, ylast);

    // --- stage 8: gate + out_proj + head (last timestep only) ---
    ycomb_kernel<<<cdiv(B_ * DI_, TB), TB, 0, stream>>>(ylast, xz, xma, mD, yc);
    mo_kernel<<<cdiv(B_ * DM_, TB), TB, 0, stream>>>(yc, mow, mo);
    pred_kernel<<<cdiv(B_ * PLNV, TB), TB, 0, stream>>>(mo, hw, hb, out);
}
